// BDH_GPU_36524401885328
// MI455X (gfx1250) — matrix operand order repaired and checked by reading, not compiled, not run on hardware
//
#include <hip/hip_runtime.h>
#include <hip/hip_bf16.h>

// ---------------------------------------------------------------------------
// BDH fast-weight RNN, MI455X (gfx1250) design:
//  * grid = 32 persistent WGs: (batch b 0..3) x (N-chunk c 0..7), 256 thr (8 waves)
//  * rho[b][:, c*128:(c+1)*128] lives in LDS for the whole T=512 scan
//    (256x128 fp32, row pitch 132 floats -> 16B-aligned float4 rows whose
//    bank-group stagger (d+vi) mod 16 gives conflict-free b128 LDS traffic)
//  * Dx/Dy/E pre-converted to bf16 + pre-swizzled into WMMA A-fragment layout
//    (init kernel) -> steady state: contiguous 32B/lane loads feeding
//    v_wmma_f32_16x16x32_bf16 (fp32 accum), prefetch next K-tile
//  * two cross-WG sync points/step: L2 float atomics + monotonic counters
// ---------------------------------------------------------------------------

typedef __attribute__((ext_vector_type(16))) __bf16 v16bf;
typedef __attribute__((ext_vector_type(8)))  float  v8f;
typedef __attribute__((ext_vector_type(4)))  float  f4;

#define B_  4
#define T_  512
#define N_  1024
#define D_  256
#define NC  128            // N-chunk per WG
#define NCHUNK (N_/NC)     // 8
#define RSTR 132           // rho row pitch (floats): 16B aligned + bank stagger
#define UDEC 0.97f
#define XDEC 0.97f
#define EPSV 1e-6f

static constexpr int    WELEMS     = N_ * D_;                 // 262144 per weight
static constexpr int    OFF_ARED   = 0;                       // float[2][B][D]
static constexpr int    OFF_VSRED  = OFF_ARED + 2*B_*D_*4;    // float[2][B][D]
static constexpr int    OFF_L1     = OFF_VSRED + 2*B_*D_*4;   // float[2][B]
static constexpr int    OFF_CNT1   = OFF_L1 + 2*B_*4;         // uint[B] monotonic
static constexpr int    OFF_CNT2   = OFF_CNT1 + B_*4;         // uint[B] monotonic
static constexpr int    OFF_WH     = ((OFF_CNT2 + B_*4 + 255)/256)*256;
static constexpr int    OFF_DXH    = OFF_WH;                  // bf16[WELEMS]
static constexpr int    OFF_DYH    = OFF_DXH + WELEMS*2;
static constexpr int    OFF_EH     = OFF_DYH + WELEMS*2;
static constexpr int    NRED_F32   = OFF_WH / 4;
static constexpr size_t SMEM_BYTES = (size_t)D_ * RSTR * 4;   // 135168 B dyn LDS

// A-fragment K permutation per ISA 16-bit A 16x32 layout:
// lane h=0 holds K 0..7,16..23 ; h=1 holds K 8..15,24..31
__device__ __host__ __forceinline__ int kmapA(int j, int h) {
  return j + ((j >> 3) << 3) + (h << 3);
}

__device__ __forceinline__ __bf16 f2bf(float f) {
  union U { float f; unsigned u; } q; q.f = f;
  unsigned r = (q.u + 0x7FFFu + ((q.u >> 16) & 1u)) >> 16;
  unsigned short hs = (unsigned short)r;
  return __builtin_bit_cast(__bf16, hs);
}

// --------------------------- init kernel -----------------------------------
// zero reduction buffers + counters; convert & swizzle Dx/Dy/E fp32 -> bf16
// WMMA-fragment order: frag = (mt*KT + kt); elem = ((frag*32 + lane)*16 + j)
__global__ void bdh_init(const float* __restrict__ Dx, const float* __restrict__ Dy,
                         const float* __restrict__ E, char* __restrict__ ws) {
  int gid = blockIdx.x * blockDim.x + threadIdx.x;
  if (gid < NRED_F32) ((float*)ws)[gid] = 0.f;   // accums + counters + pad
  if (gid < 3 * WELEMS) {
    int w = gid / WELEMS;
    int f = gid - w * WELEMS;
    const float* src = (w == 0) ? Dx : (w == 1) ? Dy : E;
    int K = (w == 2) ? N_ : D_;                  // Dx/Dy rows=n,K=d ; E rows=d,K=n
    __bf16* dst = (__bf16*)(ws + ((w == 0) ? OFF_DXH : (w == 1) ? OFF_DYH : OFF_EH));
    int j = f & 15, lane = (f >> 4) & 31, t2 = f >> 9;
    int KT = K / 32, kt = t2 % KT, mt = t2 / KT;
    int m = lane & 15, h = lane >> 4;
    int k = kt * 32 + kmapA(j, h);
    dst[f] = f2bf(src[(size_t)(mt * 16 + m) * K + k]);
  }
}

// --------------------------- reductions ------------------------------------
__device__ __forceinline__ float wave_sum(float v) {
#pragma unroll
  for (int o = 16; o; o >>= 1) v += __shfl_down(v, o, 32);
  return v;
}

__device__ float block_sum(float v, float* red_s, int tid) {
  v = wave_sum(v);
  __syncthreads();
  if ((tid & 31) == 0) red_s[tid >> 5] = v;
  __syncthreads();
  if (tid == 0) {
    float s = 0.f;
#pragma unroll
    for (int i = 0; i < 8; ++i) s += red_s[i];
    red_s[8] = s;
  }
  __syncthreads();
  float r = red_s[8];
  __syncthreads();
  return r;
}

// --------------------------- WMMA gemv -------------------------------------
// One wave computes 16 output rows (m-tile mt) of W @ act : acc holds the
// 16x16 fp32 C tile; activation replicated across the 16 B-columns so
// column 0 is the gemv result. B per-lane K order: h=0 -> K 0..7,16..23, h=1 -> 8..15,24..31.
template <int NKT>
__device__ __forceinline__ v8f wmma_gemv_acc(const __bf16* __restrict__ Wf, int KT_total,
                                             int mt, int kt0, const __bf16* acth,
                                             int lane, v8f acc) {
  const int h = lane >> 4;
#pragma unroll
  for (int kt = 0; kt < NKT; ++kt) {
    const v16bf* ap =
        (const v16bf*)(Wf + ((((size_t)mt * KT_total + (kt0 + kt)) * 32 + lane) << 4));
    if (kt + 1 < NKT)
      __builtin_prefetch(Wf + ((((size_t)mt * KT_total + (kt0 + kt + 1)) * 32 + lane) << 4), 0, 1);
    v16bf a = *ap;
    v16bf b;
    const __bf16* av = acth + kt * 32 + (h << 3);
#pragma unroll
    for (int j = 0; j < 16; ++j) b[j] = av[j + ((j >> 3) << 3)];
    acc = __builtin_amdgcn_wmma_f32_16x16x32_bf16(false, a, false, b, (short)0, acc,
                                                  false, false);
  }
  return acc;
}

// C layout: VGPR r, lanes 0-15 -> M=r, lanes 16-31 -> M=r+8; N = lane&15.
__device__ __forceinline__ void col0_to_lds(v8f acc, float* out_s, int mbase, int lane) {
  if ((lane & 15) == 0) {
    int mb = mbase + ((lane >> 4) ? 8 : 0);
#pragma unroll
    for (int r = 0; r < 8; ++r) out_s[mb + r] = acc[r];
  }
}

__device__ __forceinline__ void col0_atomic(v8f acc, float* gdst, int mbase, int lane) {
  if ((lane & 15) == 0) {
    int mb = mbase + ((lane >> 4) ? 8 : 0);
#pragma unroll
    for (int r = 0; r < 8; ++r)
      __hip_atomic_fetch_add(gdst + mb + r, acc[r], __ATOMIC_RELAXED,
                             __HIP_MEMORY_SCOPE_AGENT);
  }
}

// --------------------------- main persistent kernel ------------------------
__global__ __launch_bounds__(256, 1)
void bdh_step_kernel(const int* __restrict__ idx, const float* __restrict__ token_emb,
                     float* __restrict__ out, char* __restrict__ ws) {
  const int tid  = threadIdx.x;
  const int lane = tid & 31;
  const int wv   = tid >> 5;        // wave 0..7
  const int b    = blockIdx.x >> 3; // batch
  const int c    = blockIdx.x & 7;  // N-chunk

  extern __shared__ char dynsmem[];
  float* rho_s = (float*)dynsmem;   // [D_][RSTR] fast-weight tile, LDS-resident

  __shared__ __bf16 act_h[256];                 // bf16 activation vector (WMMA B)
  __shared__ __align__(16) float gout_s[128];   // gemv col-0 results
  __shared__ __align__(16) float x_s[128];      // normalized x_t chunk (state)
  __shared__ __align__(16) float xr_s[128];     // raw x chunk
  __shared__ float red_s[9];

  float*    A_red  = (float*)(ws + OFF_ARED);
  float*    VS_red = (float*)(ws + OFF_VSRED);
  float*    L1_red = (float*)(ws + OFF_L1);
  unsigned* CNT1   = (unsigned*)(ws + OFF_CNT1);
  unsigned* CNT2   = (unsigned*)(ws + OFF_CNT2);
  const __bf16* Dxh = (const __bf16*)(ws + OFF_DXH);
  const __bf16* Dyh = (const __bf16*)(ws + OFF_DYH);
  const __bf16* Eh  = (const __bf16*)(ws + OFF_EH);

  // vectorized zero of the resident fast-weight tile
  {
    f4* rz = (f4*)rho_s;
    for (int i = tid; i < (D_ * RSTR) / 4; i += 256) rz[i] = (f4){0.f, 0.f, 0.f, 0.f};
  }
  if (tid < NC) x_s[tid] = 0.f;
  __syncthreads();

  for (int t = 0; t < T_; ++t) {
    const int p = t & 1;
    float* Ap  = A_red  + (p * B_ + b) * D_;
    float* VSp = VS_red + (p * B_ + b) * D_;
    float* VSn = VS_red + ((1 - p) * B_ + b) * D_;

    // -- v_prev gather, layernorm(v_prev), bf16 for Dx gemv --
    int   tok = idx[b * T_ + t];
    float v   = token_emb[(size_t)tok * D_ + tid];
    act_h[tid] = f2bf(v);
    float mv  = block_sum(v, red_s, tid) * (1.f / D_);
    float dv  = v - mv;
    float s2  = block_sum(dv * dv, red_s, tid) * (1.f / (D_ - 1));
    float vln = dv / (sqrtf(s2) + EPSV);   // kept in-register for rho update

    // -- gemv1: Dx_chunk @ v_prev (wave wv -> m-tile) --
    {
      v8f acc = {0.f, 0.f, 0.f, 0.f, 0.f, 0.f, 0.f, 0.f};
      acc = wmma_gemv_acc<8>(Dxh, D_ / 32, c * 8 + wv, 0, act_h, lane, acc);
      col0_to_lds(acc, gout_s, wv * 16, lane);
    }
    __syncthreads();

    // -- x_raw chunk + L1 partial --
    float xr = 0.f;
    if (tid < NC) {
      xr = XDEC * x_s[tid] + fmaxf(gout_s[tid], 0.f);
      xr_s[tid] = xr;
    }
    float l1p = block_sum(fabsf(xr), red_s, tid);

    // -- a* partial: thread d=tid streams its rho row as float4 (ds_load_b128;
    //    bank group (d+vi)%16 -> conflict-free minimum-cycle LDS traffic) --
    float ap = 0.f;
    {
      const f4* row = (const f4*)(rho_s + tid * RSTR);
      const f4* xv  = (const f4*)xr_s;
      f4 accv = {0.f, 0.f, 0.f, 0.f};
#pragma unroll 8
      for (int vi = 0; vi < NC / 4; ++vi) accv += row[vi] * xv[vi];
      ap = accv.x + accv.y + accv.z + accv.w;
    }

    // -- phase-1 cross-WG reduction (a* + L1) --
    __hip_atomic_fetch_add(Ap + tid, ap, __ATOMIC_RELAXED, __HIP_MEMORY_SCOPE_AGENT);
    if (tid == 0) {
      __hip_atomic_fetch_add(&L1_red[p * B_ + b], l1p, __ATOMIC_RELAXED,
                             __HIP_MEMORY_SCOPE_AGENT);
      __hip_atomic_fetch_add(&CNT1[b], 1u, __ATOMIC_RELEASE, __HIP_MEMORY_SCOPE_AGENT);
      unsigned tgt = 8u * (unsigned)(t + 1);
      while (__hip_atomic_load(&CNT1[b], __ATOMIC_RELAXED, __HIP_MEMORY_SCOPE_AGENT) < tgt)
        __builtin_amdgcn_s_sleep(1);
    }
    __syncthreads();
    __builtin_amdgcn_fence(__ATOMIC_ACQUIRE, "agent");

    float l1  = L1_red[p * B_ + b];
    float inv = 1.f / (l1 + EPSV);
    float a_d = Ap[tid] * inv;
    if (tid < NC) x_s[tid] = xr_s[tid] * inv;
    // reset next-parity v* accumulator slice for step t+1 (safe: everyone passed
    // phase-2 of t-1 to get here)
    if (tid < 32)
      __hip_atomic_store(VSn + c * 32 + tid, 0.f, __ATOMIC_RELAXED,
                         __HIP_MEMORY_SCOPE_AGENT);

    // -- layernorm(a*) -> bf16 --
    float ma  = block_sum(a_d, red_s, tid) * (1.f / D_);
    float da  = a_d - ma;
    float sa  = block_sum(da * da, red_s, tid) * (1.f / (D_ - 1));
    act_h[tid] = f2bf(da / (sqrtf(sa) + EPSV));
    __syncthreads();

    // -- gemv2: Dy_chunk @ ln(a*) --
    {
      v8f acc = {0.f, 0.f, 0.f, 0.f, 0.f, 0.f, 0.f, 0.f};
      acc = wmma_gemv_acc<8>(Dyh, D_ / 32, c * 8 + wv, 0, act_h, lane, acc);
      col0_to_lds(acc, gout_s, wv * 16, lane);
    }
    __syncthreads();

    // -- y_t chunk = relu(y_core) * relu(x_t) -> bf16 --
    if (tid < NC)
      act_h[tid] = f2bf(fmaxf(gout_s[tid], 0.f) * fmaxf(x_s[tid], 0.f));
    __syncthreads();

    // -- gemv3: E[:, chunk] @ y_t ; partials atomically into VS accumulator --
    {
      v8f acc0 = {0.f, 0.f, 0.f, 0.f, 0.f, 0.f, 0.f, 0.f};
      v8f acc1 = {0.f, 0.f, 0.f, 0.f, 0.f, 0.f, 0.f, 0.f};
      acc0 = wmma_gemv_acc<4>(Eh, N_ / 32, 2 * wv,     c * 4, act_h, lane, acc0);
      acc1 = wmma_gemv_acc<4>(Eh, N_ / 32, 2 * wv + 1, c * 4, act_h, lane, acc1);
      col0_atomic(acc0, VSp, (2 * wv) * 16, lane);
      col0_atomic(acc1, VSp, (2 * wv + 1) * 16, lane);
    }

    // -- phase-2 cross-WG barrier --
    if (tid == 0) {
      __hip_atomic_fetch_add(&CNT2[b], 1u, __ATOMIC_RELEASE, __HIP_MEMORY_SCOPE_AGENT);
      unsigned tgt = 8u * (unsigned)(t + 1);
      while (__hip_atomic_load(&CNT2[b], __ATOMIC_RELAXED, __HIP_MEMORY_SCOPE_AGENT) < tgt)
        __builtin_amdgcn_s_sleep(1);
    }
    __syncthreads();
    __builtin_amdgcn_fence(__ATOMIC_ACQUIRE, "agent");

    // reset phase-1 parity-p accumulators for reuse at t+2
    if (tid < 32)
      __hip_atomic_store(Ap + c * 32 + tid, 0.f, __ATOMIC_RELAXED,
                         __HIP_MEMORY_SCOPE_AGENT);
    if (c == 0 && tid == 0)
      __hip_atomic_store(&L1_red[p * B_ + b], 0.f, __ATOMIC_RELAXED,
                         __HIP_MEMORY_SCOPE_AGENT);

    // -- v* layernorm + output (one chunk-WG per batch writes) --
    if (c == 0) {
      float vs  = VSp[tid];
      float mvs = block_sum(vs, red_s, tid) * (1.f / D_);
      float dvs = vs - mvs;
      float svs = block_sum(dvs * dvs, red_s, tid) * (1.f / (D_ - 1));
      out[((size_t)b * T_ + t) * D_ + tid] = dvs / (sqrtf(svs) + EPSV);
    }

    // -- fast-weight write: rho = U*(rho + ln(v_prev) ⊗ x_t), decay fused,
    //    float4 ds_store_b128, same conflict-free pattern --
    {
      f4* row = (f4*)(rho_s + tid * RSTR);
      const f4* xv = (const f4*)x_s;
#pragma unroll 8
      for (int vi = 0; vi < NC / 4; ++vi)
        row[vi] = UDEC * (row[vi] + vln * xv[vi]);
    }
    __syncthreads();
  }
}

// --------------------------- host launcher ---------------------------------
extern "C" void kernel_launch(void* const* d_in, const int* in_sizes, int n_in,
                              void* d_out, int out_size, void* d_ws, size_t ws_size,
                              hipStream_t stream) {
  (void)in_sizes; (void)n_in; (void)out_size; (void)ws_size;
  const int*   idx       = (const int*)d_in[0];
  const float* token_emb = (const float*)d_in[1];
  const float* E         = (const float*)d_in[2];
  const float* Dx        = (const float*)d_in[3];
  const float* Dy        = (const float*)d_in[4];
  float* out = (float*)d_out;
  char*  ws  = (char*)d_ws;

  // allow 132 KB dynamic LDS for the resident rho tile (idempotent)
  (void)hipFuncSetAttribute((const void*)bdh_step_kernel,
                            hipFuncAttributeMaxDynamicSharedMemorySize,
                            (int)SMEM_BYTES);

  // one-time (per launch) weight bf16 conversion + WMMA swizzle, ws reset
  bdh_init<<<(3 * WELEMS + 255) / 256, 256, 0, stream>>>(Dx, Dy, E, ws);
  // persistent scan kernel: 32 WGs (4 batches x 8 N-chunks), 8 waves each
  bdh_step_kernel<<<B_ * NCHUNK, 256, SMEM_BYTES, stream>>>(idx, token_emb, out, ws);
}
